// CausalTemporalConv1d_45165876085051
// MI455X (gfx1250) — matrix operand order repaired and checked by reading, not compiled, not run on hardware
//
#include <hip/hip_runtime.h>
#include <cstddef>

#define TT 4096
#define BB 8
#define CC 256
#define C2 512
#define NT 64

typedef __attribute__((ext_vector_type(16))) _Float16 v16h;
typedef __attribute__((ext_vector_type(8)))  _Float16 v8h;
typedef __attribute__((ext_vector_type(8)))  float    v8f;
typedef __attribute__((ext_vector_type(4)))  unsigned int v4u;
typedef __attribute__((ext_vector_type(8)))  int      v8i;
typedef __attribute__((ext_vector_type(4)))  int      v4i;

// ---- LDS layout (units: halfs). Strides chosen so every v8h (16B) LDS load
// is 16-byte aligned and rows land on distinct banks for 16-lane access.
// WS_STRIDE=40 is produced by the TDM itself: 32-half rows (16 DWORDs) with
// pad_interval=16DW / pad_amount=4DW appended by the DMA engine.
#define WS_STRIDE  40                          // weight tile rows [512][40]
#define XS_STRIDE  40                          // x^T tile rows    [64][40]
#define ACT_STRIDE 264                         // acts^T rows      [64][264]
#define WS_OFF    0
#define XS_OFF    (C2 * WS_STRIDE)             // 20480
#define GATE_OFF  (XS_OFF + NT * XS_STRIDE)    // 23040
#define ACT_OFF   (GATE_OFF + CC * NT)         // 39424
#define SMEM_HALFS (ACT_OFF + NT * ACT_STRIDE) // 56320
#define SMEM_BYTES (SMEM_HALFS * 2)            // 112640 B (< 320KB WGP LDS)

__device__ __forceinline__ v16h make16(v8h lo, v8h hi) {
  return __builtin_shufflevector(lo, hi, 0,1,2,3,4,5,6,7,8,9,10,11,12,13,14,15);
}
// A fragment (16x32 f16, row-major in LDS): lane (h=lane>>4, m=lane&15) needs
// halfs  K in [8h, 8h+8) and [16+8h, 16+8h+8)  -> two 16B loads at +0 / +16.
__device__ __forceinline__ v16h fragA(const _Float16* rowp /* = row + 8h */) {
  return make16(*(const v8h*)(rowp), *(const v8h*)(rowp + 16));
}
// B fragment (32x16 f16) fed from a transposed [n][k] panel: lane needs
// K in [8h, 8h+8) and [16+8h, 16+8h+8) -> two 16B loads at +0 / +16 halfs.
__device__ __forceinline__ v16h fragB(const _Float16* colp /* = row(n) + 8h */) {
  return make16(*(const v8h*)(colp), *(const v8h*)(colp + 16));
}

// ---- Tensor Data Mover: DMA a [rows x 32]-half tile (row stride 256 halfs in
// global) into LDS at byte offset lds_off, padding each 64B row with 16B so the
// LDS row stride becomes 40 halfs. Descriptor per CDNA5 ISA.
__device__ __forceinline__ void tdm_load_w(const _Float16* gsrc, unsigned lds_off,
                                           int rows) {
  unsigned long long ga = (unsigned long long)(size_t)gsrc;
  v4u g0;
  g0[0] = 1u;                                          // count=1 (valid, user)
  g0[1] = lds_off;                                     // lds_addr (bytes)
  g0[2] = (unsigned)(ga & 0xffffffffu);                // global_addr[31:0]
  g0[3] = (unsigned)((ga >> 32) & 0x1ffffffu) | (2u << 30);  // addr[56:32], type=2
  v8i g1;
  g1[0] = (1 << 16)            // data_size = 2 bytes
        | (1 << 20)            // pad_enable
        | (3 << 22)            // pad_interval: 16 DWORDs (one 32-half row)
        | (3 << 25);           // pad_amount: 4 DWORDs (8 halfs)
  g1[1] = (int)(256u << 16);                           // tensor_dim0 = 256 (lo16)
  g1[2] = (int)((unsigned)(rows & 0xffff) << 16);      // dim0 hi16=0 | tensor_dim1 lo16
  g1[3] = (int)(32u << 16);                            // dim1 hi16=0 | tile_dim0 = 32
  g1[4] = rows & 0xffff;                               // tile_dim1 = rows, tile_dim2=0
  g1[5] = 256;                                         // tensor_dim0_stride lo32
  g1[6] = 0;                                           // stride hi | dim1_stride lo
  g1[7] = 0;
  v4i g2 = (v4i)0;
  v4i g3 = (v4i)0;
#if defined(__clang_major__) && (__clang_major__ >= 23)
  v8i g4 = (v8i)0;
  __builtin_amdgcn_tensor_load_to_lds(g0, g1, g2, g3, g4, 0);
#else
  __builtin_amdgcn_tensor_load_to_lds(g0, g1, g2, g3, 0);
#endif
}

// ---------------- weight prep: rearrange + f32->f16 -------------------------
// wdil16[((i*3+tap)*512+co)*256+ci] = in_w[i][co][ci][tap]
// wrs16 [(i*512+co)*256+ci]         = rs_w[i][co][ci]   (slot 7 = rs_w_last)
__global__ void prep_weights(const float* __restrict__ in_w,
                             const float* __restrict__ rs_w,
                             const float* __restrict__ rs_wl,
                             _Float16* __restrict__ wdil,
                             _Float16* __restrict__ wrs) {
  size_t idx = (size_t)blockIdx.x * blockDim.x + threadIdx.x;
  const size_t NDIL = (size_t)8 * 3 * C2 * CC;
  if (idx < NDIL) {
    int ci = (int)(idx % CC);
    size_t r = idx / CC;
    int co = (int)(r % C2);
    size_t q = r / C2;
    int k = (int)(q % 3);
    int i = (int)(q / 3);
    wdil[idx] = (_Float16)in_w[(((size_t)i * C2 + co) * CC + ci) * 3 + k];
  }
  const size_t NRS = (size_t)7 * C2 * CC;
  if (idx < NRS) wrs[idx] = (_Float16)rs_w[idx];
  const size_t NRSL = (size_t)CC * CC;
  if (idx < NRSL) wrs[(size_t)7 * C2 * CC + idx] = (_Float16)rs_wl[idx];
}

// ---------------- naive pointwise (1x1) conv: start / cond / end ------------
__global__ void pw_kernel(const float* __restrict__ in, const float* __restrict__ w,
                          const float* __restrict__ bias, float* __restrict__ out,
                          int Cin, int Cout) {
  size_t idx = (size_t)blockIdx.x * blockDim.x + threadIdx.x;
  size_t total = (size_t)BB * Cout * TT;
  if (idx >= total) return;
  int t = (int)(idx % TT);
  size_t r = idx / TT;
  int co = (int)(r % Cout);
  int b  = (int)(r / Cout);
  const float* ip = in + ((size_t)b * Cin) * TT + t;
  const float* wp = w + (size_t)co * Cin;
  float s = bias[co];
  for (int ci = 0; ci < Cin; ++ci) s += wp[ci] * ip[(size_t)ci * TT];
  out[idx] = s;
}

// ---------------- fused WaveNet layer: dilated conv + GLU + res/skip --------
template<bool LAST>
__global__ __launch_bounds__(512)
void layer_kernel(const float* __restrict__ xin, const float* __restrict__ cnd,
                  const _Float16* __restrict__ wd16, const float* __restrict__ bd,
                  const _Float16* __restrict__ wr16, const float* __restrict__ br,
                  float* __restrict__ xout, float* __restrict__ skip,
                  int dil, int first) {
  extern __shared__ _Float16 smem[];
  _Float16* Ws    = smem + WS_OFF;    // [512][WS_STRIDE] f16 weight K-chunk (TDM)
  _Float16* XsT   = smem + XS_OFF;    // [64][XS_STRIDE]  f16 x^T K-chunk
  _Float16* Gate  = smem + GATE_OFF;  // [256][NT]        f16 gate pre-acts
  _Float16* ActsT = smem + ACT_OFF;   // [64][ACT_STRIDE] f16 acts^T (phase-2 B)

  const unsigned lds_ws = __builtin_amdgcn_groupstaticsize() + WS_OFF * 2;

  const int tid   = threadIdx.x;
  const int w     = tid >> 5;     // wave id (wave32), owns m rows [32w, 32w+32)
  const int lane  = tid & 31;
  const int h     = lane >> 4;
  const int lm    = lane & 15;
  const int bb    = blockIdx.y;
  const int tbase = blockIdx.x * NT;

  v8f acc[2][4];
  #pragma unroll
  for (int mt = 0; mt < 2; ++mt)
    #pragma unroll
    for (int nt = 0; nt < 4; ++nt)
      #pragma unroll
      for (int r = 0; r < 8; ++r) acc[mt][nt][r] = 0.f;

  // ---------------- Phase 1: a[512x64] = Wd(512x256x3) (*) x ---------------
  for (int tap = 0; tap < 3; ++tap) {
    const int toff = (tap - 2) * dil;         // causal: taps at t-2d, t-d, t
    for (int cc = 0; cc < 8; ++cc) {
      const int ci0 = cc * 32;
      // DMA weight tile [512][32] -> Ws (padded to 40-half rows) via TDM
      if (w == 0) {
        tdm_load_w(wd16 + ((size_t)tap * C2) * CC + ci0, lds_ws, C2);
      }
      // stage x^T chunk (32 ci x 64 n), f32 -> f16, branchless causal guard
      #pragma unroll
      for (int p = 0; p < 4; ++p) {
        int idx = tid + p * 512;
        int n = idx & 63, ci = idx >> 6;
        int t = tbase + n + toff;
        int tc = t < 0 ? 0 : t;
        float v = xin[((size_t)(bb * CC + ci0 + ci)) * TT + tc];
        v = (t < 0) ? 0.f : v;
        XsT[n * XS_STRIDE + ci] = (_Float16)v;
      }
      if (w == 0) __builtin_amdgcn_s_wait_tensorcnt(0);
      __syncthreads();
      #pragma unroll
      for (int mt = 0; mt < 2; ++mt) {
        v16h a = fragA(Ws + (w * 32 + mt * 16 + lm) * WS_STRIDE + 8 * h);
        #pragma unroll
        for (int nt = 0; nt < 4; ++nt) {
          v16h bm = fragB(XsT + (nt * 16 + lm) * XS_STRIDE + 8 * h);
          acc[mt][nt] = __builtin_amdgcn_wmma_f32_16x16x32_f16(
              false, a, false, bm, (short)0, acc[mt][nt], false, false);
        }
      }
      __syncthreads();
    }
  }

  // ---------------- bias + cond + GLU --------------------------------------
  // D layout: element (r, lane): m = base + 8h + r, n = ntile*16 + lm
  if (w >= 8) {            // gate half: channels 256..511
    #pragma unroll
    for (int mt = 0; mt < 2; ++mt) {
      int mb = w * 32 + mt * 16 + 8 * h;
      #pragma unroll
      for (int nt = 0; nt < 4; ++nt) {
        int n = nt * 16 + lm;
        #pragma unroll
        for (int r = 0; r < 8; ++r) {
          int m = mb + r;
          float g = acc[mt][nt][r] + bd[m] +
                    cnd[((size_t)(bb * C2 + m)) * TT + tbase + n];
          Gate[(m - CC) * NT + n] = (_Float16)g;
        }
      }
    }
  }
  __syncthreads();
  if (w < 8) {             // value half: acts = lo * sigmoid(gate)
    #pragma unroll
    for (int mt = 0; mt < 2; ++mt) {
      int mb = w * 32 + mt * 16 + 8 * h;
      #pragma unroll
      for (int nt = 0; nt < 4; ++nt) {
        int n = nt * 16 + lm;
        #pragma unroll
        for (int r = 0; r < 8; ++r) {
          int m = mb + r;
          float lo = acc[mt][nt][r] + bd[m] +
                     cnd[((size_t)(bb * C2 + m)) * TT + tbase + n];
          float gt = (float)Gate[m * NT + n];
          float av = lo / (1.f + __expf(-gt));
          ActsT[n * ACT_STRIDE + m] = (_Float16)av;   // acts^T [n][k]
        }
      }
    }
  }
  __syncthreads();

  // ---------------- Phase 2: rs[M2 x 64] = Wr(M2 x 256) * acts -------------
  #pragma unroll
  for (int mt = 0; mt < 2; ++mt)
    #pragma unroll
    for (int nt = 0; nt < 4; ++nt)
      #pragma unroll
      for (int r = 0; r < 8; ++r) acc[mt][nt][r] = 0.f;

  const int M2 = LAST ? CC : C2;
  for (int kc = 0; kc < 8; ++kc) {
    const int k0 = kc * 32;
    if (w == 0) {
      tdm_load_w(wr16 + k0, lds_ws, M2);
      __builtin_amdgcn_s_wait_tensorcnt(0);
    }
    __syncthreads();
    if (w * 32 < M2) {
      #pragma unroll
      for (int mt = 0; mt < 2; ++mt) {
        v16h a = fragA(Ws + (w * 32 + mt * 16 + lm) * WS_STRIDE + 8 * h);
        #pragma unroll
        for (int nt = 0; nt < 4; ++nt) {
          v16h bm = fragB(ActsT + (nt * 16 + lm) * ACT_STRIDE + k0 + 8 * h);
          acc[mt][nt] = __builtin_amdgcn_wmma_f32_16x16x32_f16(
              false, a, false, bm, (short)0, acc[mt][nt], false, false);
        }
      }
    }
    __syncthreads();
  }

  // ---------------- writeback: residual x and skip accumulation ------------
  if (w * 32 < M2) {
    #pragma unroll
    for (int mt = 0; mt < 2; ++mt) {
      int mb = w * 32 + mt * 16 + 8 * h;
      #pragma unroll
      for (int nt = 0; nt < 4; ++nt) {
        int n = nt * 16 + lm;
        int t = tbase + n;
        #pragma unroll
        for (int r = 0; r < 8; ++r) {
          int m = mb + r;
          float v = acc[mt][nt][r] + br[m];
          if (LAST) {                               // rs (256) -> skip += rs
            size_t off = ((size_t)(bb * CC + m)) * TT + t;
            skip[off] = skip[off] + v;
          } else if (w < 8) {                       // res half: x_out = x_in + rs
            size_t off = ((size_t)(bb * CC + m)) * TT + t;
            xout[off] = xin[off] + v;
          } else {                                  // skip half
            size_t off = ((size_t)(bb * CC + (m - CC))) * TT + t;
            skip[off] = first ? v : (skip[off] + v);
          }
        }
      }
    }
  }
}

// ---------------------------------------------------------------------------
extern "C" void kernel_launch(void* const* d_in, const int* in_sizes, int n_in,
                              void* d_out, int out_size, void* d_ws, size_t ws_size,
                              hipStream_t stream) {
  const float* inp     = (const float*)d_in[0];
  const float* cond    = (const float*)d_in[1];
  const float* start_w = (const float*)d_in[2];
  const float* start_b = (const float*)d_in[3];
  const float* cond_w  = (const float*)d_in[4];
  const float* cond_b  = (const float*)d_in[5];
  const float* in_w    = (const float*)d_in[6];
  const float* in_b    = (const float*)d_in[7];
  const float* rs_w    = (const float*)d_in[8];
  const float* rs_b    = (const float*)d_in[9];
  const float* rs_wl   = (const float*)d_in[10];
  const float* rs_bl   = (const float*)d_in[11];
  const float* end_w   = (const float*)d_in[12];
  const float* end_b   = (const float*)d_in[13];

  // workspace: x ping-pong, cond, skip (5 * B*C*T f32 = 168 MB)
  //            + f16 weights for TDM (wdil 6.3MB, wrs 2.1MB)
  const size_t XN = (size_t)BB * CC * TT;
  float* wsf  = (float*)d_ws;
  float* x0   = wsf;
  float* x1   = wsf + XN;
  float* cbuf = wsf + 2 * XN;       // B x 512 x T
  float* skip = wsf + 4 * XN;
  _Float16* wdil16 = (_Float16*)(wsf + 5 * XN);
  _Float16* wrs16  = wdil16 + (size_t)8 * 3 * C2 * CC;

  {
    size_t total = (size_t)8 * 3 * C2 * CC;   // covers all three weight blobs
    prep_weights<<<dim3((unsigned)((total + 255) / 256)), dim3(256), 0, stream>>>(
        in_w, rs_w, rs_wl, wdil16, wrs16);
  }
  {
    size_t total = (size_t)BB * CC * TT;
    pw_kernel<<<dim3((unsigned)((total + 255) / 256)), dim3(256), 0, stream>>>(
        inp, start_w, start_b, x0, 80, CC);
  }
  {
    size_t total = (size_t)BB * C2 * TT;
    pw_kernel<<<dim3((unsigned)((total + 255) / 256)), dim3(256), 0, stream>>>(
        cond, cond_w, cond_b, cbuf, CC, C2);
  }

  float* xi = x0;
  float* xo = x1;
  dim3 g(TT / NT, BB);
  for (int i = 0; i < 8; ++i) {
    const _Float16* wd = wdil16 + (size_t)i * 3 * C2 * CC;
    const float*    bd = in_b + (size_t)i * C2;
    const _Float16* wr = wrs16 + (size_t)i * C2 * CC;
    if (i < 7) {
      layer_kernel<false><<<g, dim3(512), SMEM_BYTES, stream>>>(
          xi, cbuf, wd, bd, wr, rs_b + (size_t)i * C2,
          xo, skip, 1 << i, (i == 0) ? 1 : 0);
    } else {
      layer_kernel<true><<<g, dim3(512), SMEM_BYTES, stream>>>(
          xi, cbuf, wd, bd, wr, rs_bl, xo, skip, 1 << i, 0);
    }
    float* tmp = xi; xi = xo; xo = tmp;
  }

  {
    size_t total = (size_t)BB * 80 * TT;
    pw_kernel<<<dim3((unsigned)((total + 255) / 256)), dim3(256), 0, stream>>>(
        skip, end_w, end_b, (float*)d_out, CC, 80);
  }
}
